// ScatterCombiner_4724464025619
// MI455X (gfx1250) — matrix operand order repaired and checked by reading, not compiled, not run on hardware
//
#include <hip/hip_runtime.h>

// ---------------------------------------------------------------------------
// CDNA5 (gfx1250, wave32) scatter-attention:
//   q = AQ @ Wq + bq ; k = X @ Wk + bk ; v = X @ Wv + bv
//   scores = <k, q[idx]> / 8 ; segment softmax ; applied = segsum(p*v)
//   out = applied @ Wo + bo
// GEMMs run on v_wmma_f32_16x16x32_bf16 (fp32 streamed through bf16).
// Wave tile = 32 rows x 32 cols (2x2 WMMA tiles): each B fragment feeds two
// WMMAs, halving global B-fragment loads per WMMA vs a 16x64 wave tile.
// ---------------------------------------------------------------------------

typedef __attribute__((ext_vector_type(16))) __bf16 v16bf;
typedef __attribute__((ext_vector_type(8)))  float  v8f;

union ABFrag { v16bf v; unsigned int u[8]; };

__device__ __forceinline__ unsigned int f2bf(float x) {
    // round-to-nearest-even fp32 -> bf16 (bits in low 16)
    unsigned int u = __float_as_uint(x);
    return (u + 0x7FFFu + ((u >> 16) & 1u)) >> 16;
}

__device__ __forceinline__ void atomicMaxFloat(float* addr, float v) {
    // monotone bit-pattern trick (valid for mixed-sign floats)
    if (v >= 0.0f) atomicMax((int*)addr, __float_as_int(v));
    else           atomicMin((unsigned int*)addr, __float_as_uint(v));
}

// ---------------------------------------------------------------------------
// Block GEMM: rows [row_base, row_base+32) of A (f32, 256 cols) times
// Wt (bf16 bits, layout Wt[col*256 + d], col=0..255) + bias -> ldsOut[32][256].
// 256 threads = 8 waves; wave w owns cols [w*32, w*32+32) over all 32 rows.
// ---------------------------------------------------------------------------
__device__ __forceinline__ void gemm_tile(const float* __restrict__ A,
                                          const unsigned short* __restrict__ Wt,
                                          const float* __restrict__ bias,
                                          int row_base, int M,
                                          unsigned short* ldsA, float* ldsOut) {
    const int t = threadIdx.x;

    // --- stage A tile: 32 rows x 256 f32 -> bf16 in LDS (row-major) --------
    {
        const int row = t >> 3;
        const int c0  = (t & 7) * 32;
        const int gr  = row_base + row;
        unsigned int* dst = (unsigned int*)(ldsA + row * 256 + c0);
        if (gr < M) {
            const float4* src = (const float4*)(A + (size_t)gr * 256 + c0);
#pragma unroll
            for (int j = 0; j < 8; ++j) {
                float4 f = src[j];
                dst[j * 2 + 0] = f2bf(f.x) | (f2bf(f.y) << 16);
                dst[j * 2 + 1] = f2bf(f.z) | (f2bf(f.w) << 16);
            }
        } else {
#pragma unroll
            for (int j = 0; j < 16; ++j) dst[j] = 0u;
        }
    }
    __syncthreads();

    const int lane = t & 31;
    const int wave = t >> 5;      // 0..7 : col strip (32 cols)
    const int half = lane >> 4;   // K-half selector per ISA fragment layout
    const int l16  = lane & 15;

    v8f acc[4] = {};   // [row strip s][col tile ct] = acc[s*2+ct]

#pragma unroll
    for (int kk = 0; kk < 256; kk += 32) {
        // A fragments for both 16-row strips (16x32 bf16 each):
        // lanes 0-15 hold K {0..7,16..23}, lanes 16-31 hold K {8..15,24..31}
        ABFrag a0, a1;
        {
            const unsigned short* base0 = ldsA + (0 * 16 + l16) * 256 + kk;
            const unsigned short* base1 = ldsA + (1 * 16 + l16) * 256 + kk;
            const unsigned int* p00 = (const unsigned int*)(base0 + half * 8);
            const unsigned int* p01 = (const unsigned int*)(base0 + 16 + half * 8);
            const unsigned int* p10 = (const unsigned int*)(base1 + half * 8);
            const unsigned int* p11 = (const unsigned int*)(base1 + 16 + half * 8);
#pragma unroll
            for (int j = 0; j < 4; ++j) {
                a0.u[j] = p00[j]; a0.u[4 + j] = p01[j];
                a1.u[j] = p10[j]; a1.u[4 + j] = p11[j];
            }
        }
        // B fragments for the wave's two 16-col tiles (32x16 bf16 each):
        // lane = column, lanes 0-15 hold K {0..7,16..23}, lanes 16-31 hold K {8..15,24..31}
        ABFrag b0, b1;
        {
            const int col0 = wave * 32 + 0 * 16 + l16;
            const int col1 = wave * 32 + 1 * 16 + l16;
            const unsigned int* p0 = (const unsigned int*)(Wt + (size_t)col0 * 256 + kk + half * 8);
            const unsigned int* p1 = (const unsigned int*)(Wt + (size_t)col1 * 256 + kk + half * 8);
#pragma unroll
            for (int j = 0; j < 4; ++j) { b0.u[j] = p0[j]; b0.u[4 + j] = p0[8 + j]; b1.u[j] = p1[j]; b1.u[4 + j] = p1[8 + j]; }
        }
        // 4 back-to-back WMMAs, each B fragment reused twice
        acc[0] = __builtin_amdgcn_wmma_f32_16x16x32_bf16(false, a0.v, false, b0.v, (short)0, acc[0], false, false);
        acc[2] = __builtin_amdgcn_wmma_f32_16x16x32_bf16(false, a1.v, false, b0.v, (short)0, acc[2], false, false);
        acc[1] = __builtin_amdgcn_wmma_f32_16x16x32_bf16(false, a0.v, false, b1.v, (short)0, acc[1], false, false);
        acc[3] = __builtin_amdgcn_wmma_f32_16x16x32_bf16(false, a1.v, false, b1.v, (short)0, acc[3], false, false);
    }

    // --- epilogue: C layout lane=N col, VGPR r -> row r (+8 for lanes 16-31)
#pragma unroll
    for (int s = 0; s < 2; ++s) {
#pragma unroll
        for (int ct = 0; ct < 2; ++ct) {
            const int col = wave * 32 + ct * 16 + l16;
            const float bb = bias[col];
#pragma unroll
            for (int r = 0; r < 8; ++r) {
                const int row = s * 16 + half * 8 + r;
                ldsOut[row * 256 + col] = acc[s * 2 + ct][r] + bb;
            }
        }
    }
    __syncthreads();
}

// ---------------------------------------------------------------------------
// Weight repack: Wq/Wk/Wv (H,D,K) and Wo (256,256) -> bf16, Wt[col*256 + d]
// ---------------------------------------------------------------------------
__global__ void k_prep_weights(const float* __restrict__ Wq, const float* __restrict__ Wk,
                               const float* __restrict__ Wv, const float* __restrict__ Wo,
                               unsigned short* __restrict__ Wq_t, unsigned short* __restrict__ Wk_t,
                               unsigned short* __restrict__ Wv_t, unsigned short* __restrict__ Wo_t) {
    const int i = blockIdx.x * blockDim.x + threadIdx.x;   // 0..65535
    if (i >= 256 * 256) return;
    const int m   = blockIdx.y;                            // which matrix
    const int col = i >> 8;
    const int d   = i & 255;
    float v;
    unsigned short* dst;
    if (m < 3) {
        const float* W = (m == 0) ? Wq : (m == 1) ? Wk : Wv;
        const int h = col >> 6, kq = col & 63;
        v   = W[h * 16384 + d * 64 + kq];
        dst = (m == 0) ? Wq_t : (m == 1) ? Wk_t : Wv_t;
    } else {
        v   = Wo[d * 256 + col];
        dst = Wo_t;
    }
    dst[(size_t)col * 256 + d] = (unsigned short)f2bf(v);
}

__global__ void k_init(float* __restrict__ mbuf, float* __restrict__ sbuf,
                       float* __restrict__ applied, int G) {
    const int stride = gridDim.x * blockDim.x;
    const int total  = G * 256;
    for (int i = blockIdx.x * blockDim.x + threadIdx.x; i < total; i += stride) {
        applied[i] = 0.0f;
        if (i < G * 4) { mbuf[i] = -__builtin_huge_valf(); sbuf[i] = 0.0f; }
    }
}

// Plain GEMM + bias -> out (used for q projection and final o projection)
__global__ void k_gemm_out(const float* __restrict__ A, const unsigned short* __restrict__ Wt,
                           const float* __restrict__ bias, float* __restrict__ out, int M) {
    __shared__ unsigned short ldsA[32 * 256];
    __shared__ float ldsOut[32 * 256];
    const int row_base = blockIdx.x * 32;
    gemm_tile(A, Wt, bias, row_base, M, ldsA, ldsOut);
    const int t   = threadIdx.x;
    const int row = t >> 3;
    const int c0  = (t & 7) * 32;
    const int gr  = row_base + row;
    if (gr < M) {
        float4* dst = (float4*)(out + (size_t)gr * 256 + c0);
        const float4* src = (const float4*)(ldsOut + row * 256 + c0);
#pragma unroll
        for (int j = 0; j < 8; ++j) dst[j] = src[j];
    }
}

// k projection fused with score = <k, q[idx]>/8 and segment max
__global__ void k_gemm_k_scores(const float* __restrict__ X, const unsigned short* __restrict__ Wt,
                                const float* __restrict__ bias, const float* __restrict__ q,
                                const long long* __restrict__ idx, float* __restrict__ ebuf,
                                float* __restrict__ mbuf, int M) {
    __shared__ unsigned short ldsA[32 * 256];
    __shared__ float ldsOut[32 * 256];
    const int row_base = blockIdx.x * 32;
    gemm_tile(X, Wt, bias, row_base, M, ldsA, ldsOut);
    const int t = threadIdx.x;
    if (t < 128) {
        const int r = t >> 2, h = t & 3;
        const int n = row_base + r;
        if (n < M) {
            const int g = (int)idx[n];
            const float4* kp = (const float4*)(ldsOut + r * 256 + h * 64);
            const float4* qp = (const float4*)(q + (size_t)g * 256 + h * 64);
            float acc = 0.0f;
#pragma unroll
            for (int j = 0; j < 16; ++j) {
                float4 a = kp[j], b = qp[j];
                acc += a.x * b.x + a.y * b.y + a.z * b.z + a.w * b.w;
            }
            const float sc = acc * 0.125f;   // 1/sqrt(64)
            ebuf[n * 4 + h] = sc;
            atomicMaxFloat(&mbuf[g * 4 + h], sc);
        }
    }
}

// e = exp(score - m[g]); s[g] += e  (one row of 4 heads per thread, float4)
__global__ void k_exp(const long long* __restrict__ idx, const float* __restrict__ mbuf,
                      float* __restrict__ ebuf, float* __restrict__ sbuf, int N) {
    const int stride = gridDim.x * blockDim.x;
    for (int n = blockIdx.x * blockDim.x + threadIdx.x; n < N; n += stride) {
        const int g = (int)idx[n];
        float4 e = *(float4*)(ebuf + (size_t)n * 4);
        const float4 m = *(const float4*)(mbuf + (size_t)g * 4);
        e.x = __expf(e.x - m.x);
        e.y = __expf(e.y - m.y);
        e.z = __expf(e.z - m.z);
        e.w = __expf(e.w - m.w);
        *(float4*)(ebuf + (size_t)n * 4) = e;
        atomicAdd(&sbuf[g * 4 + 0], e.x);
        atomicAdd(&sbuf[g * 4 + 1], e.y);
        atomicAdd(&sbuf[g * 4 + 2], e.z);
        atomicAdd(&sbuf[g * 4 + 3], e.w);
    }
}

// v projection fused with p*v scatter-accumulate into applied[G,256]
__global__ void k_gemm_v_apply(const float* __restrict__ X, const unsigned short* __restrict__ Wt,
                               const float* __restrict__ bias, const long long* __restrict__ idx,
                               const float* __restrict__ ebuf, const float* __restrict__ sbuf,
                               float* __restrict__ applied, int M) {
    __shared__ unsigned short ldsA[32 * 256];
    __shared__ float ldsOut[32 * 256];
    const int row_base = blockIdx.x * 32;
    gemm_tile(X, Wt, bias, row_base, M, ldsA, ldsOut);
    const int t   = threadIdx.x;
    const int r   = t >> 3;
    const int c0  = (t & 7) * 32;
    const int n   = row_base + r;
    if (n < M) {
        const int g = (int)idx[n];
        float p[4];
#pragma unroll
        for (int h = 0; h < 4; ++h) p[h] = ebuf[n * 4 + h] / sbuf[g * 4 + h];
        float* ap = applied + (size_t)g * 256;
#pragma unroll
        for (int j = 0; j < 32; ++j) {
            const int c = c0 + j;
            atomicAdd(&ap[c], p[c >> 6] * ldsOut[r * 256 + c]);
        }
    }
}

// ---------------------------------------------------------------------------
extern "C" void kernel_launch(void* const* d_in, const int* in_sizes, int n_in,
                              void* d_out, int out_size, void* d_ws, size_t ws_size,
                              hipStream_t stream) {
    const float*     X   = (const float*)d_in[0];
    const long long* idx = (const long long*)d_in[1];
    const float*     AQ  = (const float*)d_in[2];
    /* d_in[3] = dim_size scalar (derived from sizes instead) */
    const float* Wq = (const float*)d_in[4];
    const float* bq = (const float*)d_in[5];
    const float* Wk = (const float*)d_in[6];
    const float* bk = (const float*)d_in[7];
    const float* Wv = (const float*)d_in[8];
    const float* bv = (const float*)d_in[9];
    const float* Wo = (const float*)d_in[10];
    const float* bo = (const float*)d_in[11];

    const int N = in_sizes[0] / 256;
    const int G = in_sizes[2] / 256;
    float* out = (float*)d_out;

    // workspace carve-out (256B aligned)
    char* ws = (char*)d_ws;
    size_t off = 0;
    auto alloc = [&](size_t bytes) -> void* {
        void* p = ws + off;
        off += (bytes + 255) & ~(size_t)255;
        return p;
    };
    unsigned short* Wq_t = (unsigned short*)alloc(65536 * 2);
    unsigned short* Wk_t = (unsigned short*)alloc(65536 * 2);
    unsigned short* Wv_t = (unsigned short*)alloc(65536 * 2);
    unsigned short* Wo_t = (unsigned short*)alloc(65536 * 2);
    float* qbuf    = (float*)alloc((size_t)G * 256 * 4);
    float* ebuf    = (float*)alloc((size_t)N * 4 * 4);
    float* mbuf    = (float*)alloc((size_t)G * 4 * 4);
    float* sbuf    = (float*)alloc((size_t)G * 4 * 4);
    float* applied = (float*)alloc((size_t)G * 256 * 4);

    const dim3 b256(256);
    k_prep_weights<<<dim3(256, 4), b256, 0, stream>>>(Wq, Wk, Wv, Wo, Wq_t, Wk_t, Wv_t, Wo_t);
    k_init<<<dim3(1024), b256, 0, stream>>>(mbuf, sbuf, applied, G);
    k_gemm_out<<<dim3((G + 31) / 32), b256, 0, stream>>>(AQ, Wq_t, bq, qbuf, G);
    k_gemm_k_scores<<<dim3((N + 31) / 32), b256, 0, stream>>>(X, Wk_t, bk, qbuf, idx, ebuf, mbuf, N);
    k_exp<<<dim3(2048), b256, 0, stream>>>(idx, mbuf, ebuf, sbuf, N);
    k_gemm_v_apply<<<dim3((N + 31) / 32), b256, 0, stream>>>(X, Wv_t, bv, idx, ebuf, sbuf, applied, N);
    k_gemm_out<<<dim3((G + 31) / 32), b256, 0, stream>>>(applied, Wo_t, bo, out, G);
}
